// GNOBlock_11553462026776
// MI455X (gfx1250) — matrix operand order repaired and checked by reading, not compiled, not run on hardware
//
#include <hip/hip_runtime.h>
#include <hip/hip_bf16.h>

typedef __attribute__((ext_vector_type(16))) __bf16 v16bf;
typedef __attribute__((ext_vector_type(8)))  float  v8f;

union FragBF { v16bf v; uint4 q[2]; };

__device__ __forceinline__ unsigned short f2bf(float f) {
    unsigned u = __builtin_bit_cast(unsigned, f);
    u += 0x7FFFu + ((u >> 16) & 1u);            // round-to-nearest-even
    return (unsigned short)(u >> 16);
}
__device__ __forceinline__ float gelu_exact(float x) {
    return 0.5f * x * (1.0f + erff(x * 0.70710678118654752f));
}

#define TM 64   // edges per block in the MLP kernel

// ---------------------------------------------------------------------------
// Prep: W2^T, W3^T -> bf16 (so WMMA B-fragments are contiguous per lane)
// ---------------------------------------------------------------------------
__global__ __launch_bounds__(256) void gno_prep(
    const float* __restrict__ w2, const float* __restrict__ w3,
    unsigned short* __restrict__ w2t, unsigned short* __restrict__ w3t)
{
    int idx = blockIdx.x * 256 + threadIdx.x;
    if (idx < 128 * 128) {                       // W2T[n][k] = W2[k][n]
        int n = idx >> 7, k = idx & 127;
        w2t[idx] = f2bf(w2[k * 128 + n]);
    } else if (idx < 128 * 128 + 256 * 128) {    // W3T[n][k] = W3[k][n]
        int j = idx - 128 * 128;
        int n = j >> 7, k = j & 127;
        w3t[j] = f2bf(w3[k * 256 + n]);
    }
}

// ---------------------------------------------------------------------------
// Fused edge MLP: w[e] = (gelu(gelu(ea@W1+b1)@W2+b2))@W3+b3  -> bf16 [E,256]
// 64 edges per 256-thread block (8 wave32).  Layers 2/3 via bf16 WMMA.
// Single 32KB LDS buffer: cols 0..127 = h1, cols 128..255 = h2, then reused
// whole as the 64x256 bf16 output staging tile for coalesced b128 stores.
// ---------------------------------------------------------------------------
__global__ __launch_bounds__(256) void gno_edge_mlp(
    const float* __restrict__ edge_attr,
    const float* __restrict__ w1, const float* __restrict__ b1,
    const unsigned short* __restrict__ w2t, const float* __restrict__ b2,
    const unsigned short* __restrict__ w3t, const float* __restrict__ b3,
    unsigned short* __restrict__ wOut, int nEdges)
{
    __shared__ float          sEA[TM][6];
    __shared__ unsigned short sBuf[TM][256];   // 32 KB bf16 staging

    const int  tid = threadIdx.x;
    const long e0  = (long)blockIdx.x * TM;

    // ---- stage edge_attr tile ----
    for (int i = tid; i < TM * 6; i += 256) {
        long g = e0 * 6 + i;
        sEA[i / 6][i % 6] = (g < (long)nEdges * 6) ? edge_attr[g] : 0.0f;
    }
    __syncthreads();

    // ---- Layer 1 (K=6, VALU): h1 = gelu(ea @ W1 + b1) -> sBuf cols 0..127 ----
    {
        const int col  = tid & 127;
        const int rb   = (tid >> 7) * 32;
        float wc[6];
        #pragma unroll
        for (int i = 0; i < 6; ++i) wc[i] = w1[i * 128 + col];
        const float bb = b1[col];
        for (int rr = 0; rr < 32; ++rr) {
            const int r = rb + rr;
            float acc = bb;
            #pragma unroll
            for (int i = 0; i < 6; ++i) acc += sEA[r][i] * wc[i];
            sBuf[r][col] = f2bf(gelu_exact(acc));
        }
    }
    __syncthreads();

    const int wave  = tid >> 5;
    const int lane  = tid & 31;
    const int lhalf = lane >> 4;     // 0 or 1
    const int l16   = lane & 15;
    const int rbase = (wave & 3) * 16;

    // ---- Layer 2 (WMMA bf16): h2 = gelu(h1 @ W2 + b2) -> sBuf cols 128..255 ----
    {
        FragBF a[4];
        #pragma unroll
        for (int k = 0; k < 4; ++k) {        // A 16x32 frag: lane<16 -> K 0..7 & 16..23
            const int kk = k * 32 + lhalf * 8;
            a[k].q[0] = *(const uint4*)&sBuf[rbase + l16][kk];
            a[k].q[1] = *(const uint4*)&sBuf[rbase + l16][kk + 16];
        }
        const int cg = (wave >> 2) * 4;
        for (int ct = 0; ct < 4; ++ct) {
            const int nrow = (cg + ct) * 16 + l16;
            v8f acc = {};
            #pragma unroll
            for (int k = 0; k < 4; ++k) {    // B 32x16 frag: lane<16 -> K 0..7 & 16..23 of col n
                FragBF bfr;
                const unsigned short* bp = w2t + nrow * 128 + k * 32 + lhalf * 8;
                bfr.q[0] = *(const uint4*)bp;
                bfr.q[1] = *(const uint4*)(bp + 16);
                acc = __builtin_amdgcn_wmma_f32_16x16x32_bf16(
                          false, a[k].v, false, bfr.v, (short)0, acc, false, false);
            }
            const float bb = b2[nrow];
            #pragma unroll
            for (int r = 0; r < 8; ++r) {    // C/D: VGPR r -> M=r+8*lhalf, N=l16
                sBuf[rbase + r + lhalf * 8][128 + nrow] = f2bf(gelu_exact(acc[r] + bb));
            }
        }
    }
    __syncthreads();

    // ---- Layer 3 (WMMA bf16): w = h2 @ W3 + b3 -> sBuf (full 64x256 tile) ----
    {
        FragBF a[4];
        #pragma unroll
        for (int k = 0; k < 4; ++k) {
            const int kk = k * 32 + lhalf * 8;
            a[k].q[0] = *(const uint4*)&sBuf[rbase + l16][128 + kk];
            a[k].q[1] = *(const uint4*)&sBuf[rbase + l16][128 + kk + 16];
        }
        __syncthreads();   // everyone has A frags in regs; sBuf is now free
        const int cg = (wave >> 2) * 8;
        for (int ct = 0; ct < 8; ++ct) {
            const int nrow = (cg + ct) * 16 + l16;
            v8f acc = {};
            #pragma unroll
            for (int k = 0; k < 4; ++k) {
                FragBF bfr;
                const unsigned short* bp = w3t + nrow * 128 + k * 32 + lhalf * 8;
                bfr.q[0] = *(const uint4*)bp;
                bfr.q[1] = *(const uint4*)(bp + 16);
                acc = __builtin_amdgcn_wmma_f32_16x16x32_bf16(
                          false, a[k].v, false, bfr.v, (short)0, acc, false, false);
            }
            const float bb = b3[nrow];
            #pragma unroll
            for (int r = 0; r < 8; ++r) {
                sBuf[rbase + r + lhalf * 8][nrow] = f2bf(acc[r] + bb);
            }
        }
    }
    __syncthreads();

    // ---- coalesced tile flush: 64x256 bf16 = 32KB via global_store_b128 ----
    {
        const uint4* s4 = (const uint4*)&sBuf[0][0];
        uint4*       d4 = (uint4*)(wOut + e0 * 256);
        if (e0 + TM <= nEdges) {
            #pragma unroll
            for (int j = 0; j < (TM * 256) / (8 * 256); ++j)   // 8 chunks/thread
                d4[tid + j * 256] = s4[tid + j * 256];
        } else {
            for (int j = tid; j < (TM * 256) / 8; j += 256)
                if (e0 + (long)(j / 32) < nEdges) d4[j] = s4[j];
        }
    }
}

// ---------------------------------------------------------------------------
// Message passing: msg[e] = x[src[e]] @ w[e]; atomic scatter-add into agg[dst]
// 8 lanes per edge, 2 packed bf16 output columns per lane (coalesced 32B rows)
// w (164MB bf16) is L2-resident (192MB), so reads run at L2 bandwidth.
// ---------------------------------------------------------------------------
__global__ __launch_bounds__(256) void gno_message(
    const unsigned short* __restrict__ wOut,
    const float* __restrict__ x,
    const int* __restrict__ eidx,          // [2, E]
    float* __restrict__ agg, int nEdges)
{
    const int  tid  = threadIdx.x;
    const int  slot = tid >> 3;            // 32 edges per block
    const int  l    = tid & 7;             // column pair 2l, 2l+1
    const long e    = (long)blockIdx.x * 32 + slot;
    if (e >= nEdges) return;

    const int src = eidx[e];
    const int dst = eidx[nEdges + e];
    const unsigned* wr = (const unsigned*)(wOut + e * 256);
    const float*    xp = x + (long)src * 16;

    float a0 = 0.f, a1 = 0.f;
    #pragma unroll
    for (int i = 0; i < 16; ++i) {
        const unsigned pk = wr[i * 8 + l];             // w[e][16*i + 2l .. +1]
        const float    xv = xp[i];
        a0 += xv * __builtin_bit_cast(float, pk << 16);
        a1 += xv * __builtin_bit_cast(float, pk & 0xFFFF0000u);
    }
    atomicAdd(&agg[(long)dst * 16 + 2 * l],     a0);
    atomicAdd(&agg[(long)dst * 16 + 2 * l + 1], a1);
}

// ---------------------------------------------------------------------------
// Node update: out = agg + x @ roots[p] + bias[p]  (gelu on pass 0)
// ---------------------------------------------------------------------------
__global__ __launch_bounds__(256) void gno_update(
    const float* __restrict__ agg, const float* __restrict__ x,
    const float* __restrict__ roots, const float* __restrict__ bias,
    float* __restrict__ out, int nNodes, int applyGelu)
{
    const long t = (long)blockIdx.x * 256 + threadIdx.x;
    if (t >= (long)nNodes * 16) return;
    const int n = (int)(t >> 4);
    const int c = (int)(t & 15);
    float acc = agg[t] + bias[c];
    const float* xp = x + (long)n * 16;
    #pragma unroll
    for (int i = 0; i < 16; ++i) acc += xp[i] * roots[i * 16 + c];
    out[t] = applyGelu ? gelu_exact(acc) : acc;
}

// ---------------------------------------------------------------------------
extern "C" void kernel_launch(void* const* d_in, const int* in_sizes, int n_in,
                              void* d_out, int out_size, void* d_ws, size_t ws_size,
                              hipStream_t stream) {
    const float* nodes     = (const float*)d_in[0];
    const float* edge_attr = (const float*)d_in[1];
    const float* k_w1      = (const float*)d_in[2];
    const float* k_b1      = (const float*)d_in[3];
    const float* k_w2      = (const float*)d_in[4];
    const float* k_b2      = (const float*)d_in[5];
    const float* k_w3      = (const float*)d_in[6];
    const float* k_b3      = (const float*)d_in[7];
    const float* roots     = (const float*)d_in[8];   // [2,16,16]
    const float* biases    = (const float*)d_in[9];   // [2,16]
    const int*   eidx      = (const int*)  d_in[10];  // [2,E]

    const int E = in_sizes[10] / 2;                   // 320000
    const int N = in_sizes[0]  / 16;                  // 20000

    // workspace layout (256B aligned)
    auto al = [](size_t x) { return (x + 255) & ~(size_t)255; };
    char* ws = (char*)d_ws;
    size_t off = 0;
    unsigned short* wOut = (unsigned short*)(ws + off); off = al(off + (size_t)E * 256 * 2);
    unsigned short* w2t  = (unsigned short*)(ws + off); off = al(off + 128 * 128 * 2);
    unsigned short* w3t  = (unsigned short*)(ws + off); off = al(off + 256 * 128 * 2);
    float*          x1   = (float*)(ws + off);         off = al(off + (size_t)N * 16 * 4);
    float*          agg  = (float*)(ws + off);         off = al(off + (size_t)N * 16 * 4);
    (void)ws_size; (void)n_in; (void)out_size;

    // 1) transpose/convert weights to bf16
    gno_prep<<<(128 * 128 + 256 * 128 + 255) / 256, 256, 0, stream>>>(k_w2, k_w3, w2t, w3t);

    // 2) edge kernel MLP -> per-edge 16x16 matrices (bf16), WMMA GEMMs
    gno_edge_mlp<<<(E + TM - 1) / TM, 256, 0, stream>>>(
        edge_attr, k_w1, k_b1, w2t, k_b2, w3t, k_b3, wOut, E);

    // 3) pass 0: x1 = gelu(scatter(msg) + nodes@roots0 + b0)
    hipMemsetAsync(agg, 0, (size_t)N * 16 * 4, stream);
    gno_message<<<(E + 31) / 32, 256, 0, stream>>>(wOut, nodes, eidx, agg, E);
    gno_update<<<(N * 16 + 255) / 256, 256, 0, stream>>>(
        agg, nodes, roots, biases, x1, N, 1);

    // 4) pass 1: out = scatter(msg) + x1@roots1 + b1   (no gelu)
    hipMemsetAsync(agg, 0, (size_t)N * 16 * 4, stream);
    gno_message<<<(E + 31) / 32, 256, 0, stream>>>(wOut, x1, eidx, agg, E);
    gno_update<<<(N * 16 + 255) / 256, 256, 0, stream>>>(
        agg, x1, roots + 256, biases + 16, (float*)d_out, N, 0);
}
